// TrafficPredictorGNN_76519137345540
// MI455X (gfx1250) — matrix operand order repaired and checked by reading, not compiled, not run on hardware
//
#include <hip/hip_runtime.h>
#include <hip/hip_bf16.h>

typedef __attribute__((ext_vector_type(8)))  _Float16 v8h;
typedef __attribute__((ext_vector_type(16))) _Float16 v16h;
typedef __attribute__((ext_vector_type(8)))  float    v8f;

#define NNODES 50000
#define HDIM   128

// ---------------- utility kernels ----------------

__global__ void k_zero_f32(float* p, int n) {
    int i = blockIdx.x * blockDim.x + threadIdx.x;
    if (i < n) p[i] = 0.0f;
}

__global__ void k_cast_f32_to_f16(const float* __restrict__ in, _Float16* __restrict__ out, int n) {
    int i = blockIdx.x * blockDim.x + threadIdx.x;
    if (i < n) out[i] = (_Float16)in[i];
}

// deg accumulation: dinv_buf[dst[e]] += ew[e]
__global__ void k_deg_add(const int* __restrict__ dst, const float* __restrict__ ew,
                          float* __restrict__ deg, int E) {
    int e = blockIdx.x * blockDim.x + threadIdx.x;
    if (e < E) unsafeAtomicAdd(&deg[dst[e]], ew[e]);
}

// dinv[i] = rsqrt(deg[i] + 1)   (deg includes self-loop weight 1)
__global__ void k_deg_finish(float* __restrict__ dinv, int n) {
    int i = blockIdx.x * blockDim.x + threadIdx.x;
    if (i < n) dinv[i] = __frsqrt_rn(dinv[i] + 1.0f);
}

// ---------------- WMMA GEMM:  C[N,128] = A16[N,128] @ W[128,128] ----------------
// Pre-swizzle W (f32 row-major [K=128][N=128]) into per-lane contiguous B fragments (f16).
// Fragment (t,j): K base 32t, N base 16j.  Lane l<16 holds col 16j+l, K=32t+e (e<8), 32t+16+(e-8) (e>=8);
// lane l>=16 holds col 16j+(l-16), K=32t+8+e (e<8), 32t+24+(e-8) (e>=8).  Storage: Bfrag[t*4096 + j*512 + lane*16 + e].
__global__ void k_prep_B(const float* __restrict__ W, _Float16* __restrict__ Bfrag) {
    int tid = blockIdx.x * blockDim.x + threadIdx.x;  // 16384 threads
    int e    = tid & 15;
    int lane = (tid >> 4) & 31;
    int j    = (tid >> 9) & 7;
    int t    = tid >> 12;
    int k = 32 * t + ((lane & 16) ? 8 : 0) + (e & 7) + ((e & 8) ? 16 : 0);
    int n = 16 * j + (lane & 15);
    Bfrag[tid] = (_Float16)W[k * HDIM + n];
}

// 256 threads = 8 waves; block handles 16 rows x 128 cols; wave j owns cols [16j,16j+16)
__global__ void __launch_bounds__(256)
k_gemm_wmma(const _Float16* __restrict__ A16, const _Float16* __restrict__ Bfrag,
            float* __restrict__ C) {
    const int lane  = threadIdx.x & 31;
    const int j     = threadIdx.x >> 5;        // wave id = N tile
    const int mbase = blockIdx.x * 16;
    const int row   = mbase + (lane & 15);
    const int khalf = (lane >> 4) * 8;         // 0 or 8

    v8f acc = {};
#pragma unroll
    for (int t = 0; t < 4; ++t) {
        // A fragment: elements 0..7 = K[32t+khalf .. +7], 8..15 = K[32t+16+khalf .. +7]
        const v8h a0 = *(const v8h*)(A16 + row * HDIM + 32 * t + khalf);
        const v8h a1 = *(const v8h*)(A16 + row * HDIM + 32 * t + 16 + khalf);
        v16h a;
#pragma unroll
        for (int i = 0; i < 8; ++i) { a[i] = a0[i]; a[i + 8] = a1[i]; }
        const v16h b = *(const v16h*)(Bfrag + (t * 8 + j) * 512 + lane * 16);
        acc = __builtin_amdgcn_wmma_f32_16x16x32_f16(
            /*neg_a=*/false, a, /*neg_b=*/false, b,
            /*c_mod=*/(short)0, acc, /*reuse_a=*/false, /*reuse_b=*/false);
    }
    // C layout: VGPR v -> M = v + 8*(lane>=16), N = lane&15 (within tile)
    const int col   = 16 * j + (lane & 15);
    const int rbase = mbase + ((lane >> 4) * 8);
#pragma unroll
    for (int v = 0; v < 8; ++v) C[(rbase + v) * HDIM + col] = acc[v];
}

// ---------------- edge scatter: agg[dst] += xw[src] * (dinv[src]*ew*dinv[dst]) ----------------
// one wave per edge; each lane handles 4 consecutive features (float4)
__global__ void __launch_bounds__(256)
k_edge_scatter(const int* __restrict__ src, const int* __restrict__ dst,
               const float* __restrict__ ew, const float* __restrict__ dinv,
               const float* __restrict__ xw, float* __restrict__ agg, int E) {
    const int e = blockIdx.x * 8 + (threadIdx.x >> 5);
    if (e >= E) return;
    const int s = src[e];
    const int d = dst[e];
    const float nrm = dinv[s] * ew[e] * dinv[d];
    const int lane = threadIdx.x & 31;
    const float4 v = *(const float4*)(xw + s * HDIM + lane * 4);
    float* p = agg + d * HDIM + lane * 4;
    unsafeAtomicAdd(p + 0, v.x * nrm);
    unsafeAtomicAdd(p + 1, v.y * nrm);
    unsafeAtomicAdd(p + 2, v.z * nrm);
    unsafeAtomicAdd(p + 3, v.w * nrm);
}

// ---------------- combine: h = ELU(agg + xw*dinv^2 + b) (+ residual); emit f32 + optional f16 ----------------
__global__ void __launch_bounds__(256)
k_combine(const float* __restrict__ agg, const float* __restrict__ xw,
          const float* __restrict__ dinv, const float* __restrict__ bias,
          const float* __restrict__ hres,   // residual (may be null)
          float* __restrict__ hout, _Float16* __restrict__ hout16, int write16, int n) {
    int i = blockIdx.x * blockDim.x + threadIdx.x;
    if (i >= n) return;
    const int row = i >> 7;
    const int col = i & 127;
    const float di = dinv[row];
    float h = agg[i] + xw[i] * di * di + bias[col];
    h = (h > 0.0f) ? h : (expf(h) - 1.0f);           // ELU(alpha=1)
    if (hres) h += hres[i];
    hout[i] = h;
    if (write16) hout16[i] = (_Float16)h;
}

// ---------------- final head: out = sigmoid(h3 @ Wl + bl), one wave per node ----------------
__global__ void __launch_bounds__(256)
k_final(const float* __restrict__ h3, const float* __restrict__ Wl,
        const float* __restrict__ bl, float* __restrict__ out, int n) {
    const int node = blockIdx.x * 8 + (threadIdx.x >> 5);
    if (node >= n) return;
    const int lane = threadIdx.x & 31;
    const float4 hv = *(const float4*)(h3 + node * HDIM + lane * 4);
    const float4 wv = *(const float4*)(Wl + lane * 4);
    float p = hv.x * wv.x + hv.y * wv.y + hv.z * wv.z + hv.w * wv.w;
#pragma unroll
    for (int off = 16; off > 0; off >>= 1) p += __shfl_xor(p, off, 32);
    if (lane == 0) out[node] = 1.0f / (1.0f + expf(-(p + bl[0])));
}

// ---------------- host orchestration ----------------

static inline size_t alignup(size_t x) { return (x + 255) & ~size_t(255); }

extern "C" void kernel_launch(void* const* d_in, const int* in_sizes, int n_in,
                              void* d_out, int out_size, void* d_ws, size_t ws_size,
                              hipStream_t stream) {
    const float* x   = (const float*)d_in[0];
    const int*   ei  = (const int*)d_in[1];
    const float* ew  = (const float*)d_in[2];
    const float* W1  = (const float*)d_in[3];
    const float* b1  = (const float*)d_in[4];
    const float* W2  = (const float*)d_in[5];
    const float* b2  = (const float*)d_in[6];
    const float* W3  = (const float*)d_in[7];
    const float* b3  = (const float*)d_in[8];
    const float* Wl  = (const float*)d_in[9];
    const float* bl  = (const float*)d_in[10];

    const int N  = in_sizes[0] / HDIM;   // 50000
    const int E  = in_sizes[2];          // 800000
    const int NH = N * HDIM;
    const int* src = ei;
    const int* dst = ei + E;

    // workspace layout
    char* ws = (char*)d_ws;
    size_t off = 0;
    float*    dinv  = (float*)(ws + off); off += alignup(size_t(N) * 4);
    float*    agg   = (float*)(ws + off); off += alignup(size_t(NH) * 4);
    float*    xw    = (float*)(ws + off); off += alignup(size_t(NH) * 4);
    float*    hA    = (float*)(ws + off); off += alignup(size_t(NH) * 4);
    float*    hB    = (float*)(ws + off); off += alignup(size_t(NH) * 4);
    _Float16* A16   = (_Float16*)(ws + off); off += alignup(size_t(NH) * 2);
    _Float16* Bfrag = (_Float16*)(ws + off); off += alignup(size_t(HDIM) * HDIM * 2);
    (void)ws_size;

    const int TB = 256;
    const int gN    = (N + TB - 1) / TB;
    const int gNH   = (NH + TB - 1) / TB;
    const int gE    = (E + TB - 1) / TB;
    const int gEdge = (E + 7) / 8;       // 1 wave per edge
    const int gGemm = N / 16;            // 3125
    const int gWave = (N + 7) / 8;

    // 1) degrees -> dinv
    k_zero_f32<<<gN, TB, 0, stream>>>(dinv, N);
    k_deg_add<<<gE, TB, 0, stream>>>(dst, ew, dinv, E);
    k_deg_finish<<<gN, TB, 0, stream>>>(dinv, N);

    // 2) layer 1: input x
    k_cast_f32_to_f16<<<gNH, TB, 0, stream>>>(x, A16, NH);
    k_prep_B<<<64, TB, 0, stream>>>(W1, Bfrag);
    k_gemm_wmma<<<gGemm, TB, 0, stream>>>(A16, Bfrag, xw);
    k_zero_f32<<<gNH, TB, 0, stream>>>(agg, NH);
    k_edge_scatter<<<gEdge, TB, 0, stream>>>(src, dst, ew, dinv, xw, agg, E);
    k_combine<<<gNH, TB, 0, stream>>>(agg, xw, dinv, b1, nullptr, hA, A16, 1, NH);

    // 3) layer 2: input h1 (A16), residual hA -> hB
    k_prep_B<<<64, TB, 0, stream>>>(W2, Bfrag);
    k_gemm_wmma<<<gGemm, TB, 0, stream>>>(A16, Bfrag, xw);
    k_zero_f32<<<gNH, TB, 0, stream>>>(agg, NH);
    k_edge_scatter<<<gEdge, TB, 0, stream>>>(src, dst, ew, dinv, xw, agg, E);
    k_combine<<<gNH, TB, 0, stream>>>(agg, xw, dinv, b2, hA, hB, A16, 1, NH);

    // 4) layer 3: input h2 (A16), residual hB -> hA
    k_prep_B<<<64, TB, 0, stream>>>(W3, Bfrag);
    k_gemm_wmma<<<gGemm, TB, 0, stream>>>(A16, Bfrag, xw);
    k_zero_f32<<<gNH, TB, 0, stream>>>(agg, NH);
    k_edge_scatter<<<gEdge, TB, 0, stream>>>(src, dst, ew, dinv, xw, agg, E);
    k_combine<<<gNH, TB, 0, stream>>>(agg, xw, dinv, b3, hB, hA, A16, 0, NH);

    // 5) head
    k_final<<<gWave, TB, 0, stream>>>(hA, Wl, bl, (float*)d_out, N);
}
